// QuantumModel_22497038696963
// MI455X (gfx1250) — matrix operand order repaired and checked by reading, not compiled, not run on hardware
//
#include <hip/hip_runtime.h>

// CDNA5 / gfx1250: wave32, WMMA f16->f32, 320KB LDS per WGP.
// One sample per workgroup; full 2^16 complex state resident in LDS as
// packed (re,im) f16x2 words (256KB). Gate math via v_wmma_f32_16x16x32_f16
// with the complex dimension interleaved along K.

typedef __attribute__((ext_vector_type(16))) _Float16 v16h;
typedef __attribute__((ext_vector_type(8)))  float    v8f;

#define NQ      16
#define NL      4
#define NSTATE  65536      // 2^16 amplitudes
#define THREADS 512        // 16 waves of 32

union V16HU { v16h v; _Float16 a[16]; unsigned u[8]; };
union V8F   { v8f  v; float    a[8]; };
union H2    { unsigned u; _Float16 h[2]; };

// LDS layout (bytes)
#define OFF_SC    0          // 65536 * u32 (packed re,im f16) = 262144
#define OFF_UR    262144     // 256 * f32
#define OFF_UI    263168     // 256 * f32
#define OFF_EVS   264192     // 16 * f32
#define OFF_XROW  264256     // 16 * f32
#define OFF_WTS   264320     // 192 * f32
#define SMEM_BYTES 265088

// Build fused 4-wire 16x16 unitary (Kronecker of four 2x2s) into LDS.
// Group covers flat-index bits b..b+3; wire w maps to bit (15-w).
// mode 0: RX(x[wire]) embedding; mode 1: Rot(phi,theta,omega) from wl.
__device__ __forceinline__ void build_U(float* Ur, float* Ui, int b, int mode,
                                        const float* xrow, const float* wl) {
  int t = threadIdx.x;
  if (t < 256) {
    int i = t >> 4, j = t & 15;
    float ar = 1.f, ai = 0.f;
#pragma unroll
    for (int k = 0; k < 4; ++k) {
      int wq = 15 - (b + k);              // wire for bit b+k
      int ik = (i >> k) & 1, jk = (j >> k) & 1;
      float ur, ui;
      if (mode == 0) {
        float h = 0.5f * xrow[wq];
        float c = __cosf(h), s = __sinf(h);
        if (ik == jk) { ur = c;   ui = 0.f; }
        else          { ur = 0.f; ui = -s;  }   // RX off-diagonal = -i*sin
      } else {
        float phi = wl[wq * 3 + 0], th = wl[wq * 3 + 1], om = wl[wq * 3 + 2];
        float ct = __cosf(0.5f * th), st = __sinf(0.5f * th);
        float ap = 0.5f * (phi + om), am = 0.5f * (phi - om);
        if      (ik == 0 && jk == 0) { ur =  __cosf(ap) * ct; ui = -__sinf(ap) * ct; }
        else if (ik == 0 && jk == 1) { ur = -__cosf(am) * st; ui = -__sinf(am) * st; }
        else if (ik == 1 && jk == 0) { ur =  __cosf(am) * st; ui = -__sinf(am) * st; }
        else                         { ur =  __cosf(ap) * ct; ui =  __sinf(ap) * ct; }
      }
      float nr = ar * ur - ai * ui;
      float ni = ar * ui + ai * ur;
      ar = nr; ai = ni;
    }
    Ur[t] = ar; Ui[t] = ai;
  }
}

// Apply 16x16 complex unitary on bits b..b+3 of the packed LDS state.
// Complex GEMM with complex interleaved along K (K=32 = 16 j x {re,im}):
//   A_re[i][2j,2j+1] = ( Ur, -Ui ),  A_im[i][2j,2j+1] = ( Ui, Ur )
//   B[2j][n] = re_j,  B[2j+1][n] = im_j   (one packed b32 word per j)
// b is a compile-time constant so b==0 coalesces to ds_load/store_b128.
template <int b>
__device__ __forceinline__ void apply_U(unsigned* sc, const float* Ur, const float* Ui) {
  const int lane = threadIdx.x & 31;
  const int wv   = threadIdx.x >> 5;
  const int m    = lane & 15;   // A row (i) / B,D column (n)
  const int kh   = lane >> 4;   // K-half select

  // A fragments: contiguous K — lane<16: K 0..15; lane>=16: K 16..31 (matches B).
  V16HU are, aim;
#pragma unroll
  for (int e = 0; e < 16; ++e) {
    int v = e >> 1;
    int k = e + kh * 16;
    int j = k >> 1, c = k & 1;
    if (c == 0) {
      are.a[e] = (_Float16)( Ur[m * 16 + j]);
      aim.a[e] = (_Float16)( Ui[m * 16 + j]);
    } else {
      are.a[e] = (_Float16)(-Ui[m * 16 + j]);
      aim.a[e] = (_Float16)( Ur[m * 16 + j]);
    }
  }

  const unsigned omask = (1u << b) - 1u;
  const int jbase = kh * 8;   // lanes 0-15: j=0..7 (K 0..15); lanes 16-31: j=8..15
  const v8f cz = {};

  const int tiles_per_wave = 256 / (THREADS / 32);
  for (int t = wv * tiles_per_wave; t < (wv + 1) * tiles_per_wave; ++t) {
    unsigned o = (unsigned)t * 16u + (unsigned)m;            // column (other-index)
    unsigned colbase = ((o >> b) << (b + 4)) + (o & omask);  // word index of column base
    V16HU bf;
#pragma unroll
    for (int e = 0; e < 8; ++e)                              // packed (re,im) word per j
      bf.u[e] = sc[colbase + ((unsigned)(jbase + e) << b)];

    V8F dre, dim;
    dre.v = __builtin_amdgcn_wmma_f32_16x16x32_f16(false, are.v, false, bf.v,
                                                   (short)0, cz, false, false);
    dim.v = __builtin_amdgcn_wmma_f32_16x16x32_f16(false, aim.v, false, bf.v,
                                                   (short)0, cz, false, false);

#pragma unroll
    for (int r = 0; r < 8; ++r) {
      unsigned mm = (unsigned)r + (unsigned)kh * 8u;          // D row M
      H2 h;
      h.h[0] = (_Float16)dre.a[r];
      h.h[1] = (_Float16)dim.a[r];
      sc[colbase + (mm << b)] = h.u;
    }
  }
}

template <int b>
__device__ __forceinline__ void do_group(unsigned* sc, float* Ur, float* Ui,
                                         int mode, const float* xrow, const float* wl) {
  build_U(Ur, Ui, b, mode, xrow, wl);
  __syncthreads();
  apply_U<b>(sc, Ur, Ui);
  __syncthreads();
}

__global__ __launch_bounds__(THREADS, 1)
void qsim_kernel(const float* __restrict__ x, const float* __restrict__ w,
                 float* __restrict__ out) {
  extern __shared__ char smem[];
  unsigned* sc   = (unsigned*)(smem + OFF_SC);
  float*    Ur   = (float*)(smem + OFF_UR);
  float*    Ui   = (float*)(smem + OFF_UI);
  float*    evs  = (float*)(smem + OFF_EVS);
  float*    xrow = (float*)(smem + OFF_XROW);
  float*    wts  = (float*)(smem + OFF_WTS);

  const int tid = threadIdx.x;
  const int bs  = blockIdx.x;

  if (tid < 16)  xrow[tid] = x[bs * 16 + tid];
  if (tid < 192) wts[tid]  = w[tid];

  // |0...0> : packed (1.0h, 0.0h) at index 0
  for (int i = tid; i < NSTATE; i += THREADS)
    sc[i] = (i == 0) ? 0x00003C00u : 0u;
  __syncthreads();

  // AngleEmbedding: 16 per-sample RX, fused 4 wires per 16x16 WMMA gate
  do_group<0 >(sc, Ur, Ui, 0, xrow, wts);
  do_group<4 >(sc, Ur, Ui, 0, xrow, wts);
  do_group<8 >(sc, Ur, Ui, 0, xrow, wts);
  do_group<12>(sc, Ur, Ui, 0, xrow, wts);

  // StronglyEntanglingLayers
  for (int l = 0; l < NL; ++l) {
    const float* wl = wts + l * 48;
    do_group<0 >(sc, Ur, Ui, 1, xrow, wl);
    do_group<4 >(sc, Ur, Ui, 1, xrow, wl);
    do_group<8 >(sc, Ur, Ui, 1, xrow, wl);
    do_group<12>(sc, Ur, Ui, 1, xrow, wl);

    // ring of CNOTs with range r (applied in order: they share wires)
    int r = l % (NQ - 1) + 1;
    for (int wq = 0; wq < NQ; ++wq) {
      int cb = 15 - wq;
      int tb = 15 - ((wq + r) % NQ);
      int hi = cb > tb ? cb : tb;
      int lo = cb > tb ? tb : cb;
      for (int mi = tid; mi < 16384; mi += THREADS) {
        unsigned low  = (unsigned)mi & ((1u << lo) - 1u);
        unsigned mid  = ((unsigned)mi >> lo) & ((1u << (hi - lo - 1)) - 1u);
        unsigned high = (unsigned)mi >> (hi - 1);
        unsigned base = (high << (hi + 1)) | (mid << (lo + 1)) | low;
        unsigned i1 = base | (1u << cb);          // control=1, target=0
        unsigned i2 = i1 | (1u << tb);            // target=1
        unsigned tmp = sc[i1]; sc[i1] = sc[i2]; sc[i2] = tmp;
      }
      __syncthreads();
    }
  }

  // <Z_w> = sum (+/-) |amp|^2, sign by bit (15-w)
  if (tid < 16) evs[tid] = 0.f;
  __syncthreads();
  float acc[16];
#pragma unroll
  for (int q = 0; q < 16; ++q) acc[q] = 0.f;
  for (int i = tid; i < NSTATE; i += THREADS) {
    H2 h; h.u = sc[i];
    float re = (float)h.h[0], im = (float)h.h[1];
    float p = re * re + im * im;
#pragma unroll
    for (int q = 0; q < 16; ++q)
      acc[q] += ((i >> (15 - q)) & 1) ? -p : p;
  }
#pragma unroll
  for (int q = 0; q < 16; ++q) atomicAdd(&evs[q], acc[q]);
  __syncthreads();
  if (tid < 16) out[bs * 16 + tid] = evs[tid];
}

extern "C" void kernel_launch(void* const* d_in, const int* in_sizes, int n_in,
                              void* d_out, int out_size, void* d_ws, size_t ws_size,
                              hipStream_t stream) {
  const float* x = (const float*)d_in[0];   // (B, 16) f32
  const float* w = (const float*)d_in[1];   // (4, 16, 3) f32
  float* out = (float*)d_out;               // (B, 16) f32
  int nb = in_sizes[0] / NQ;                // batch = 512
  qsim_kernel<<<nb, THREADS, SMEM_BYTES, stream>>>(x, w, out);
}
